// ControllerNetwork_58445914964562
// MI455X (gfx1250) — matrix operand order repaired and checked by reading, not compiled, not run on hardware
//
#include <hip/hip_runtime.h>
#include <hip/hip_bf16.h>

// Problem dims
#define HIDDEN  2048
#define GATES   (4 * HIDDEN)   // 8192
#define NSTEPS  8
#define NCHOICE 10

// ---- WMMA vector types (gfx1250, wave32) ----
typedef __attribute__((ext_vector_type(8)))  float   v8f;
typedef __attribute__((ext_vector_type(16))) __bf16  v16bf;
typedef __attribute__((ext_vector_type(8)))  __bf16  v8bf;

// fp32 -> bf16, round-to-nearest-even
__device__ __forceinline__ unsigned short f2bf(float f) {
    unsigned int u = __float_as_uint(f);
    u += 0x7FFFu + ((u >> 16) & 1u);
    return (unsigned short)(u >> 16);
}

__device__ __forceinline__ float sigm(float x) { return 1.0f / (1.0f + __expf(-x)); }

// Persistent per-launch state in workspace
struct LstmState {
    float h0[HIDDEN], c0[HIDDEN], h1[HIDDEN], c1[HIDDEN];
    unsigned short h0b[HIDDEN], h1b[HIDDEN];   // bf16 copies for WMMA B operand
    float x;                                    // scalar LSTM input for next step
};

// ---------------- weight conversion: fp32 -> bf16 row-major ----------------
__global__ __launch_bounds__(256)
void convert_bf16_kernel(const float* __restrict__ src,
                         unsigned short* __restrict__ dst, int n4) {
    int i = blockIdx.x * blockDim.x + threadIdx.x;
    if (i >= n4) return;
    float4 v = ((const float4*)src)[i];
    ushort4 o;
    o.x = f2bf(v.x); o.y = f2bf(v.y); o.z = f2bf(v.z); o.w = f2bf(v.w);
    ((ushort4*)dst)[i] = o;
}

// ---------------- zero the recurrent state ----------------
__global__ __launch_bounds__(256)
void init_state_kernel(LstmState* st) {
    int i = blockIdx.x * blockDim.x + threadIdx.x;
    if (i < HIDDEN) {
        st->h0[i] = 0.f; st->c0[i] = 0.f;
        st->h1[i] = 0.f; st->c1[i] = 0.f;
        st->h0b[i] = 0;  st->h1b[i] = 0;
    }
    if (i == 0) st->x = 0.f;
}

// ---------------- WMMA GEMV core ----------------
// Accumulates a 16-row tile of W[GATES x HIDDEN] (bf16, row-major) times vector
// hb[HIDDEN] (bf16).  B is broadcast into all 16 columns, so every output
// column of the 16x16 tile holds the same GEMV result.
// A layout (16-bit A 16x32): lane L holds row (L&15); half-wave 0 holds
// K = {k0..k0+7, k0+16..k0+23}, half-wave 1 holds K = {k0+8..k0+15, k0+24..k0+31}
// => two contiguous 16-byte loads per lane per K-tile.
__device__ __forceinline__ v8f wmma_gemv_accum(
        const unsigned short* __restrict__ Wb,
        const unsigned short* __restrict__ hb,
        int row0, int lane, v8f acc) {
    const int half = lane >> 4;     // 0 or 1
    const int m    = lane & 15;     // row within tile
    const unsigned short* wrow = Wb + (size_t)(row0 + m) * HIDDEN + half * 8;
    const unsigned short* bptr = hb + half * 8;    // B half-wave K base
    #pragma unroll 4
    for (int k0 = 0; k0 < HIDDEN; k0 += 32) {
        v8bf alo = *(const v8bf*)(wrow + k0);        // K = kbase .. kbase+7
        v8bf ahi = *(const v8bf*)(wrow + k0 + 16);   // K = kbase+16 .. kbase+23
        v16bf a = __builtin_shufflevector(alo, ahi,
            0, 1, 2, 3, 4, 5, 6, 7, 8, 9, 10, 11, 12, 13, 14, 15);
        v8bf blo = *(const v8bf*)(bptr + k0);
        v8bf bhi = *(const v8bf*)(bptr + k0 + 16);
        v16bf b = __builtin_shufflevector(blo, bhi,
            0, 1, 2, 3, 4, 5, 6, 7, 8, 9, 10, 11, 12, 13, 14, 15);
        acc = __builtin_amdgcn_wmma_f32_16x16x32_bf16(
            false, a, false, b, (short)0, acc, false, false);
    }
    return acc;
}

// Epilogue helper: columns are identical; C/D layout => VGPR r holds row
// (half*8 + r) for lanes 0 / 16.
#define GEMV_TILE_SETUP()                                              \
    const int lane = threadIdx.x & 31;                                 \
    const int wave = (int)((blockIdx.x * blockDim.x + threadIdx.x) >> 5); \
    const int row0 = wave * 16;

// ---------------- layer-0 gate GEMV: gates = W_hh0 @ h0 + w_ih0*x + b ------
__global__ __launch_bounds__(128)
void lstm0_gemv_kernel(const unsigned short* __restrict__ Wb_hh0,
                       const LstmState* __restrict__ st,
                       const float* __restrict__ w_ih0,
                       const float* __restrict__ b_ih0,
                       const float* __restrict__ b_hh0,
                       float* __restrict__ gates) {
    GEMV_TILE_SETUP();
    v8f acc = {};
    acc = wmma_gemv_accum(Wb_hh0, st->h0b, row0, lane, acc);
    if ((lane & 15) == 0) {
        float x = st->x;
        int base = row0 + (lane >> 4) * 8;
        #pragma unroll
        for (int r = 0; r < 8; ++r) {
            int row = base + r;
            gates[row] = acc[r] + w_ih0[row] * x + b_ih0[row] + b_hh0[row];
        }
    }
}

// ---------------- layer-1 gate GEMV: gates = W_ih1@h0 + W_hh1@h1 + b -------
__global__ __launch_bounds__(128)
void lstm1_gemv_kernel(const unsigned short* __restrict__ Wb_ih1,
                       const unsigned short* __restrict__ Wb_hh1,
                       const LstmState* __restrict__ st,
                       const float* __restrict__ b_ih1,
                       const float* __restrict__ b_hh1,
                       float* __restrict__ gates) {
    GEMV_TILE_SETUP();
    v8f acc = {};
    acc = wmma_gemv_accum(Wb_ih1, st->h0b, row0, lane, acc);
    acc = wmma_gemv_accum(Wb_hh1, st->h1b, row0, lane, acc);
    if ((lane & 15) == 0) {
        int base = row0 + (lane >> 4) * 8;
        #pragma unroll
        for (int r = 0; r < 8; ++r) {
            int row = base + r;
            gates[row] = acc[r] + b_ih1[row] + b_hh1[row];
        }
    }
}

// ---------------- LSTM cell pointwise (pytorch gate order i,f,g,o) ---------
__global__ __launch_bounds__(256)
void cell_kernel(const float* __restrict__ gates,
                 float* __restrict__ h, float* __restrict__ c,
                 unsigned short* __restrict__ hb) {
    int i = blockIdx.x * blockDim.x + threadIdx.x;
    if (i >= HIDDEN) return;
    float ig = sigm(gates[i]);
    float fg = sigm(gates[i + HIDDEN]);
    float gg = tanhf(gates[i + 2 * HIDDEN]);
    float og = sigm(gates[i + 3 * HIDDEN]);
    float cn = fg * c[i] + ig * gg;
    float hn = og * tanhf(cn);
    c[i] = cn;
    h[i] = hn;
    hb[i] = f2bf(hn);
}

// ---------------- head: logits, softmax, sample, next input ---------------
__global__ __launch_bounds__(256)
void head_kernel(const LstmState* __restrict__ st_in,
                 LstmState* __restrict__ st_out,
                 const float* __restrict__ hw,   // [NCHOICE x HIDDEN] this step
                 const float* __restrict__ hb,   // [NCHOICE] this step
                 const int* __restrict__ wtu,    // which_to_use
                 int t, float* __restrict__ out) {
    __shared__ float red[256];
    __shared__ float logits[NCHOICE];
    int tid = threadIdx.x;
    const float* h1 = st_in->h1;
    for (int cch = 0; cch < NCHOICE; ++cch) {
        float s = 0.f;
        for (int k = tid; k < HIDDEN; k += 256)
            s += hw[cch * HIDDEN + k] * h1[k];
        red[tid] = s;
        __syncthreads();
        for (int off = 128; off > 0; off >>= 1) {
            if (tid < off) red[tid] += red[tid + off];
            __syncthreads();
        }
        if (tid == 0) logits[cch] = red[0] + hb[cch];
        __syncthreads();
    }
    if (tid == 0) {
        float mx = logits[0];
        for (int cch = 1; cch < NCHOICE; ++cch) mx = fmaxf(mx, logits[cch]);
        float probs[NCHOICE];
        float denom = 0.f;
        for (int cch = 0; cch < NCHOICE; ++cch) {
            probs[cch] = __expf(logits[cch] - mx);
            denom += probs[cch];
        }
        float inv = 1.0f / denom;
        for (int cch = 0; cch < NCHOICE; ++cch) probs[cch] *= inv;
        // deterministic counter-based RNG (stand-in for jax threefry key 42)
        unsigned long long s =
            (0x9E3779B97F4A7C15ull * (unsigned long long)(t + 1)) ^ 42ull;
        s ^= s >> 33; s *= 0xFF51AFD7ED558CCDull;
        s ^= s >> 33; s *= 0xC4CEB9FE1A85EC53ull;
        s ^= s >> 33;
        float u = (float)(s >> 40) * (1.0f / 16777216.0f);
        int samp = NCHOICE - 1;
        float csum = 0.f;
        for (int cch = 0; cch < NCHOICE; ++cch) {
            csum += probs[cch];
            if (u < csum) { samp = cch; break; }
        }
        float proba = probs[samp];
        int w = wtu[t];
        int idx = (w != -1) ? w : samp;
        ((int*)out)[t] = idx;                              // idxs (int32)
        out[NSTEPS + t] = proba;                           // probas
        for (int cch = 0; cch < NCHOICE; ++cch)            // all_probs
            out[2 * NSTEPS + t * NCHOICE + cch] = probs[cch];
        st_out->x = (float)idx;                            // next LSTM input
    }
}

// ---------------- launch ----------------
// Workspace layout (needs ~100.8 MB):
//   [0)                3 bf16 matrices of GATES*HIDDEN each (w_hh0, w_ih1, w_hh1)
//   [+3*32MB)          gates0[GATES] f32, gates1[GATES] f32
//   [+64KB)           LstmState
extern "C" void kernel_launch(void* const* d_in, const int* in_sizes, int n_in,
                              void* d_out, int out_size, void* d_ws, size_t ws_size,
                              hipStream_t stream) {
    const int*   wtu    = (const int*)  d_in[0];
    const float* w_ih0  = (const float*)d_in[1];
    const float* w_hh0  = (const float*)d_in[2];
    const float* b_ih0  = (const float*)d_in[3];
    const float* b_hh0  = (const float*)d_in[4];
    const float* w_ih1  = (const float*)d_in[5];
    const float* w_hh1  = (const float*)d_in[6];
    const float* b_ih1  = (const float*)d_in[7];
    const float* b_hh1  = (const float*)d_in[8];
    const float* head_w = (const float*)d_in[9];
    const float* head_b = (const float*)d_in[10];
    float* out = (float*)d_out;

    const size_t matElems = (size_t)GATES * HIDDEN;         // 16.78M
    unsigned short* Wb_hh0 = (unsigned short*)d_ws;
    unsigned short* Wb_ih1 = Wb_hh0 + matElems;
    unsigned short* Wb_hh1 = Wb_ih1 + matElems;
    float* gates0 = (float*)(Wb_hh1 + matElems);
    float* gates1 = gates0 + GATES;
    LstmState* st = (LstmState*)(gates1 + GATES);

    // 1) zero recurrent state (workspace is poisoned, not zeroed)
    init_state_kernel<<<(HIDDEN + 255) / 256, 256, 0, stream>>>(st);

    // 2) fp32 -> bf16 weight conversion (once per launch; L2-resident after)
    int n4 = (int)(matElems / 4);
    int cgrid = (n4 + 255) / 256;
    convert_bf16_kernel<<<cgrid, 256, 0, stream>>>(w_hh0, Wb_hh0, n4);
    convert_bf16_kernel<<<cgrid, 256, 0, stream>>>(w_ih1, Wb_ih1, n4);
    convert_bf16_kernel<<<cgrid, 256, 0, stream>>>(w_hh1, Wb_hh1, n4);

    // 3) 8 sequential steps; phase ordering via stream
    const int gemvBlocks = GATES / 16 / 4;   // 512 tiles / 4 waves per block
    for (int t = 0; t < NSTEPS; ++t) {
        lstm0_gemv_kernel<<<gemvBlocks, 128, 0, stream>>>(
            Wb_hh0, st, w_ih0, b_ih0, b_hh0, gates0);
        cell_kernel<<<HIDDEN / 256, 256, 0, stream>>>(
            gates0, st->h0, st->c0, st->h0b);
        lstm1_gemv_kernel<<<gemvBlocks, 128, 0, stream>>>(
            Wb_ih1, Wb_hh1, st, b_ih1, b_hh1, gates1);
        cell_kernel<<<HIDDEN / 256, 256, 0, stream>>>(
            gates1, st->h1, st->c1, st->h1b);
        head_kernel<<<1, 256, 0, stream>>>(
            st, st, head_w + (size_t)t * NCHOICE * HIDDEN,
            head_b + t * NCHOICE, wtu, t, out);
    }
}
